// PredRNN_1829656068683
// MI455X (gfx1250) — matrix operand order repaired and checked by reading, not compiled, not run on hardware
//
#include <hip/hip_runtime.h>
#include <hip/hip_bf16.h>
#include <math.h>

#define BB 32
#define TT 13
#define WWD 11
#define HH 64
#define PIX (WWD*WWD)          /* 121  */
#define NPIX (BB*PIX)          /* 3872 = 242*16 = 121*32 */
#define STEPS (TT-1)           /* 12   */
#define GENN (BB*STEPS*PIX)    /* 46464 */

typedef __attribute__((ext_vector_type(16))) __bf16 v16bf;
typedef __attribute__((ext_vector_type(8)))  float  v8f;
typedef __attribute__((ext_vector_type(4)))  unsigned int u32x4;

__device__ __forceinline__ unsigned short f2bf(float f) {
  unsigned int u = __float_as_uint(f);
  u += 0x7FFFu + ((u >> 16) & 1u);          // round-to-nearest-even
  return (unsigned short)(u >> 16);
}
__device__ __forceinline__ float sigf(float x) { return 1.0f / (1.0f + __expf(-x)); }

// ---------------- utility kernels ----------------
__global__ void k_zero(float* p, long n) {
  long i = (long)blockIdx.x * blockDim.x + threadIdx.x;
  long s = (long)gridDim.x * blockDim.x;
  for (; i < n; i += s) p[i] = 0.0f;
}

// fp32 weight [Ksrc,N] -> bf16 TRANSPOSED [N,KP] with zero K-padding
__global__ void k_cvtw(const float* __restrict__ src, unsigned short* __restrict__ dst,
                       int Ksrc, int KP, int N) {
  long n = (long)KP * N;
  long i = (long)blockIdx.x * blockDim.x + threadIdx.x;
  long s = (long)gridDim.x * blockDim.x;
  for (; i < n; i += s) {
    int k = (int)(i % KP);
    int c = (int)(i / KP);
    dst[i] = (k < Ksrc) ? f2bf(src[(long)k * N + c]) : (unsigned short)0;
  }
}

// im2col for 5x5 SAME conv: src [B(sStride), W, W, C] f32 -> dst [NPIX, KP] bf16
__global__ void k_im2col(const float* __restrict__ src, unsigned short* __restrict__ dst,
                         int C, int KP, int sStride) {
  long n = (long)NPIX * KP;
  long i = (long)blockIdx.x * blockDim.x + threadIdx.x;
  long s = (long)gridDim.x * blockDim.x;
  int Ksrc = 25 * C;
  for (; i < n; i += s) {
    int kp = (int)(i % KP);
    int m  = (int)(i / KP);
    float v = 0.0f;
    if (kp < Ksrc) {
      int ci = kp % C;
      int kk = kp / C;
      int b = m / PIX, p = m % PIX;
      int y = p / WWD + kk / 5 - 2;
      int x = p % WWD + kk % 5 - 2;
      if ((unsigned)y < (unsigned)WWD && (unsigned)x < (unsigned)WWD)
        v = src[(long)b * sStride + (long)(y * WWD + x) * C + ci];
    }
    dst[i] = f2bf(v);
  }
}

// ---------------- WMMA GEMM: C[M,N] = A[M,KP] * Bt[N,KP]^T + bias ----------------
// block = 128 threads (4 wave32). Each wave: two 16-row M subtiles x one 16-col N
// subtile sharing the B fragment. No LDS; fragments come straight from L2 via
// per-lane global_load_b128. Ping-pong double buffering (two fragment sets, no
// rotation copies): load set-B(k+32), wmma set-A, load set-A(k+64), wmma set-B.
__global__ void __launch_bounds__(128)
k_gemm_wmma(const unsigned short* __restrict__ A, const unsigned short* __restrict__ Bt,
            const float* __restrict__ bias, float* __restrict__ Cout, int N, int KP) {
  const int lane = threadIdx.x & 31;
  const int wave = threadIdx.x >> 5;
  const int m0   = blockIdx.x * 32;                 // two 16-row subtiles per wave
  const int hi   = lane >> 4;                       // 0 / 1
  const int r16  = lane & 15;
  const int n    = blockIdx.y * 64 + (wave << 4) + r16;

  // A fragment per lane: element e<8 -> k = hi*8+e ; e>=8 -> k = 16+hi*8+(e-8)
  const unsigned short* a0 = A + (long)(m0 + r16) * KP + hi * 8;
  const unsigned short* a1 = a0 + (long)16 * KP;
  // B fragment per lane: element e<8 -> k = hi*8+e ; e>=8 -> k = 16+hi*8+(e-8)
  const unsigned short* bp = Bt + (long)n * KP + hi * 8;

  v8f acc0 = {0.f, 0.f, 0.f, 0.f, 0.f, 0.f, 0.f, 0.f};
  v8f acc1 = {0.f, 0.f, 0.f, 0.f, 0.f, 0.f, 0.f, 0.f};

  union Frag { u32x4 q[2]; v16bf v; };
  Frag A0a, A1a, Ba, A0b, A1b, Bb;

  auto LD = [&](Frag& f0, Frag& f1, Frag& f2, int kc) {
    f0.q[0] = *(const u32x4*)(a0 + kc);
    f0.q[1] = *(const u32x4*)(a0 + kc + 16);
    f1.q[0] = *(const u32x4*)(a1 + kc);
    f1.q[1] = *(const u32x4*)(a1 + kc + 16);
    f2.q[0] = *(const u32x4*)(bp + kc);
    f2.q[1] = *(const u32x4*)(bp + kc + 16);
  };

  LD(A0a, A1a, Ba, 0);                    // prologue: chunk 0
  int kc = 0;
  for (; kc + 64 <= KP; kc += 64) {       // two chunks per iteration
    LD(A0b, A1b, Bb, kc + 32);
    __builtin_prefetch(a0 + kc + 64, 0, 3);   // gfx1250 global_prefetch
    __builtin_prefetch(a1 + kc + 64, 0, 3);
    __builtin_prefetch(bp + kc + 64, 0, 3);
    acc0 = __builtin_amdgcn_wmma_f32_16x16x32_bf16(false, A0a.v, false, Ba.v,
                                                   (short)0, acc0, false, false);
    acc1 = __builtin_amdgcn_wmma_f32_16x16x32_bf16(false, A1a.v, false, Ba.v,
                                                   (short)0, acc1, false, false);
    if (kc + 64 < KP) LD(A0a, A1a, Ba, kc + 64);
    acc0 = __builtin_amdgcn_wmma_f32_16x16x32_bf16(false, A0b.v, false, Bb.v,
                                                   (short)0, acc0, false, false);
    acc1 = __builtin_amdgcn_wmma_f32_16x16x32_bf16(false, A1b.v, false, Bb.v,
                                                   (short)0, acc1, false, false);
  }
  if (kc < KP) {                          // odd tail chunk (KP == 32 case)
    acc0 = __builtin_amdgcn_wmma_f32_16x16x32_bf16(false, A0a.v, false, Ba.v,
                                                   (short)0, acc0, false, false);
    acc1 = __builtin_amdgcn_wmma_f32_16x16x32_bf16(false, A1a.v, false, Ba.v,
                                                   (short)0, acc1, false, false);
  }

  const float bv = bias ? bias[n] : 0.0f;
  const int mb0 = m0 + hi * 8;
  #pragma unroll
  for (int r = 0; r < 8; r++) {
    Cout[(long)(mb0 + r) * N + n]      = acc0[r] + bv;
    Cout[(long)(mb0 + 16 + r) * N + n] = acc1[r] + bv;
  }
}

// ---------------- LayerNorm over (W,W,C) per batch sample, in place ----------------
__global__ void k_ln(float* __restrict__ buf, const float* __restrict__ g,
                     const float* __restrict__ be, int C) {
  const int b = blockIdx.x;
  const int n = PIX * C;
  float* x = buf + (long)b * n;
  float s = 0.f, s2 = 0.f;
  for (int i = threadIdx.x; i < n; i += blockDim.x) { float v = x[i]; s += v; s2 += v * v; }
  __shared__ float rs[256], rq[256];
  rs[threadIdx.x] = s; rq[threadIdx.x] = s2;
  __syncthreads();
  for (int o = 128; o > 0; o >>= 1) {
    if (threadIdx.x < o) { rs[threadIdx.x] += rs[threadIdx.x + o]; rq[threadIdx.x] += rq[threadIdx.x + o]; }
    __syncthreads();
  }
  const float mean = rs[0] / n;
  const float var  = rq[0] / n - mean * mean;
  const float inv  = rsqrtf(var + 1e-5f);
  for (int i = threadIdx.x; i < n; i += blockDim.x) {
    float v = x[i]; int ch = i % C;
    x[i] = (v - mean) * inv * g[ch] + be[ch];
  }
}

// ---------------- gate kernels ----------------
__global__ void k_gates1(const float* __restrict__ xcc, const float* __restrict__ hcc,
                         const float* __restrict__ ccc, float* __restrict__ c, long n) {
  long i = (long)blockIdx.x * blockDim.x + threadIdx.x;
  long s = (long)gridDim.x * blockDim.x;
  for (; i < n; i += s) {
    long m = i / HH; int ch = (int)(i % HH);
    const float* xr = xcc + m * 448;
    const float* hr = hcc + m * 256;
    const float* cr = ccc + m * 192;
    float iv = sigf (xr[ch]       + hr[ch]       + cr[ch]);
    float gv = tanhf(xr[64 + ch]  + hr[64 + ch]  + cr[64 + ch]);
    float fv = sigf (xr[128 + ch] + hr[128 + ch] + cr[128 + ch] + 1.0f);
    c[i] = fv * c[i] + iv * gv;
  }
}
__global__ void k_gates2(const float* __restrict__ c2m, const float* __restrict__ xcc,
                         const float* __restrict__ mcc, float* __restrict__ mb, long n) {
  long i = (long)blockIdx.x * blockDim.x + threadIdx.x;
  long s = (long)gridDim.x * blockDim.x;
  for (; i < n; i += s) {
    long m = i / HH; int ch = (int)(i % HH);
    const float* cr = c2m + m * 256;
    const float* xr = xcc + m * 448;
    const float* mr = mcc + m * 192;
    float ii = sigf (cr[ch]       + xr[256 + ch] + mr[ch]);
    float gg = tanhf(cr[64 + ch]  + xr[320 + ch]);
    float ff = sigf (cr[128 + ch] + xr[384 + ch] + mr[64 + ch] + 1.0f);
    mb[i] = ff * tanhf(mr[128 + ch]) + ii * gg;
  }
}
__global__ void k_gates3(const float* __restrict__ xcc, const float* __restrict__ hcc,
                         const float* __restrict__ c2m, const float* __restrict__ om,
                         const float* __restrict__ cell, float* __restrict__ h, long n) {
  long i = (long)blockIdx.x * blockDim.x + threadIdx.x;
  long s = (long)gridDim.x * blockDim.x;
  for (; i < n; i += s) {
    long m = i / HH; int ch = (int)(i % HH);
    float o = tanhf(xcc[m * 448 + 192 + ch] + hcc[m * 256 + 192 + ch] +
                    c2m[m * 256 + 192 + ch] + om[i]);
    h[i] = o * tanhf(cell[i]);
  }
}
__global__ void k_concat_bf(const float* __restrict__ c, const float* __restrict__ m,
                            unsigned short* __restrict__ dst, long n) {
  long i = (long)blockIdx.x * blockDim.x + threadIdx.x;
  long s = (long)gridDim.x * blockDim.x;
  for (; i < n; i += s) {
    long mm = i / HH; int ch = (int)(i % HH);
    dst[mm * 128 + ch]      = f2bf(c[i]);
    dst[mm * 128 + 64 + ch] = f2bf(m[i]);
  }
}
__global__ void k_ghu(const float* __restrict__ zcc, const float* __restrict__ xgcc,
                      float* __restrict__ z, long n) {
  long i = (long)blockIdx.x * blockDim.x + threadIdx.x;
  long s = (long)gridDim.x * blockDim.x;
  for (; i < n; i += s) {
    long m = i / HH; int ch = (int)(i % HH);
    float pg = tanhf(zcc[m * 128 + ch]      + xgcc[m * 128 + ch]);
    float u  = sigf (zcc[m * 128 + 64 + ch] + xgcc[m * 128 + 64 + ch]);
    z[i] = u * pg + (1.0f - u) * z[i];
  }
}

// 1x1 out conv (K=64, N=1) + scatter into gen_images slice t
__global__ void k_outconv(const float* __restrict__ h2, const float* __restrict__ w,
                          const float* __restrict__ b, float* __restrict__ gen, int t) {
  int m = blockIdx.x * blockDim.x + threadIdx.x;
  if (m >= NPIX) return;
  const float* hr = h2 + (long)m * HH;
  float acc = b[0];
  #pragma unroll 8
  for (int k = 0; k < HH; k++) acc += hr[k] * w[k];
  int bb = m / PIX, p = m % PIX;
  gen[((long)bb * STEPS + t) * PIX + p] = acc;
}

__global__ void k_loss_zero(float* loss) { if (threadIdx.x == 0 && blockIdx.x == 0) *loss = 0.0f; }

__global__ void k_loss(const float* __restrict__ gen, const float* __restrict__ images,
                       float* __restrict__ loss) {
  float s = 0.0f;
  for (long i = (long)blockIdx.x * blockDim.x + threadIdx.x; i < GENN;
       i += (long)gridDim.x * blockDim.x) {
    long b = i / (STEPS * PIX);
    long r = i % (STEPS * PIX);
    long t = r / PIX, p = r % PIX;
    float d = gen[i] - images[(b * TT + t + 1) * PIX + p];
    s += 0.5f * d * d;
  }
  __shared__ float rs[256];
  rs[threadIdx.x] = s; __syncthreads();
  for (int o = 128; o > 0; o >>= 1) {
    if (threadIdx.x < o) rs[threadIdx.x] += rs[threadIdx.x + o];
    __syncthreads();
  }
  if (threadIdx.x == 0) atomicAdd(loss, rs[0]);
}

// ---------------- host orchestration ----------------
extern "C" void kernel_launch(void* const* d_in, const int* in_sizes, int n_in,
                              void* d_out, int out_size, void* d_ws, size_t ws_size,
                              hipStream_t stream) {
  (void)in_sizes; (void)n_in; (void)out_size;
  const float* IM = (const float*)d_in[0];
  auto LP = [&](int l, int j) -> const float* { return (const float*)d_in[2 + 26 * l + j]; };
  auto GP = [&](int j)        -> const float* { return (const float*)d_in[80 + j]; };
  const float* OUT_W = (const float*)d_in[88];
  const float* OUT_B = (const float*)d_in[89];
  float* gen  = (float*)d_out;
  float* loss = gen + GENN;

  char* ws = (char*)d_ws;
  size_t off = 0;
  auto alloc = [&](size_t bytes) -> void* {
    void* p = ws + off;
    off = (off + bytes + 255) & ~(size_t)255;
    return p;
  };

  // bf16 weights (transposed [N,KP])
  unsigned short *wx_bf[3], *wh_bf[3], *wc_bf[3], *wm_bf[3], *wc2m_bf[3], *wom_bf[3], *wcell_bf[3];
  const int KPX[3] = {32, 1600, 1600};
  const int KSX[3] = {25, 1600, 1600};
  for (int l = 0; l < 3; l++) {
    wx_bf[l]    = (unsigned short*)alloc((size_t)KPX[l] * 448 * 2);
    wh_bf[l]    = (unsigned short*)alloc((size_t)1600 * 256 * 2);
    wc_bf[l]    = (unsigned short*)alloc((size_t)1600 * 192 * 2);
    wm_bf[l]    = (unsigned short*)alloc((size_t)1600 * 192 * 2);
    wc2m_bf[l]  = (unsigned short*)alloc((size_t)1600 * 256 * 2);
    wom_bf[l]   = (unsigned short*)alloc((size_t)1600 * 64 * 2);
    wcell_bf[l] = (unsigned short*)alloc((size_t)128 * 64 * 2);
  }
  unsigned short* wz_bf  = (unsigned short*)alloc((size_t)1600 * 128 * 2);
  unsigned short* wxg_bf = (unsigned short*)alloc((size_t)1600 * 128 * 2);

  // state + intermediate buffers (f32)
  float* hbuf[3]; float* cbuf[3];
  for (int l = 0; l < 3; l++) {
    hbuf[l] = (float*)alloc((size_t)NPIX * HH * 4);
    cbuf[l] = (float*)alloc((size_t)NPIX * HH * 4);
  }
  float* mbuf  = (float*)alloc((size_t)NPIX * HH * 4);
  float* zbuf  = (float*)alloc((size_t)NPIX * HH * 4);
  float* xcc   = (float*)alloc((size_t)NPIX * 448 * 4);
  float* hcc   = (float*)alloc((size_t)NPIX * 256 * 4);
  float* ccc   = (float*)alloc((size_t)NPIX * 192 * 4);
  float* mcc   = (float*)alloc((size_t)NPIX * 192 * 4);
  float* c2m   = (float*)alloc((size_t)NPIX * 256 * 4);
  float* om    = (float*)alloc((size_t)NPIX * 64 * 4);
  float* cellb = (float*)alloc((size_t)NPIX * 64 * 4);
  float* zcc   = (float*)alloc((size_t)NPIX * 128 * 4);
  float* xgcc  = (float*)alloc((size_t)NPIX * 128 * 4);
  unsigned short* colbuf = (unsigned short*)alloc((size_t)NPIX * 1600 * 2);
  unsigned short* catbuf = (unsigned short*)alloc((size_t)NPIX * 128 * 2);
  if (off > ws_size) return;  // workspace too small: no-op

  auto cvt = [&](const float* s, unsigned short* d, int Ks, int KP, int N) {
    long n = (long)KP * N;
    int blocks = (int)((n + 255) / 256);
    k_cvtw<<<blocks, 256, 0, stream>>>(s, d, Ks, KP, N);
  };
  for (int l = 0; l < 3; l++) {
    cvt(LP(l, 0),  wx_bf[l],    KSX[l], KPX[l], 448);
    cvt(LP(l, 4),  wh_bf[l],    1600, 1600, 256);
    cvt(LP(l, 8),  wc_bf[l],    1600, 1600, 192);
    cvt(LP(l, 12), wm_bf[l],    1600, 1600, 192);
    cvt(LP(l, 16), wc2m_bf[l],  1600, 1600, 256);
    cvt(LP(l, 20), wom_bf[l],   1600, 1600, 64);
    cvt(LP(l, 24), wcell_bf[l], 128, 128, 64);
  }
  cvt(GP(0), wz_bf,  1600, 1600, 128);
  cvt(GP(4), wxg_bf, 1600, 1600, 128);

  const long NG = (long)NPIX * HH;
  auto zero = [&](float* p, long n) { k_zero<<<512, 256, 0, stream>>>(p, n); };
  for (int l = 0; l < 3; l++) { zero(hbuf[l], NG); zero(cbuf[l], NG); }
  zero(mbuf, NG); zero(zbuf, NG);
  k_loss_zero<<<1, 64, 0, stream>>>(loss);

  auto conv_ln = [&](const float* src, int C, int sStride, const unsigned short* Wb, int KP,
                     const float* bias, int N, float* outb, const float* g, const float* be) {
    long tot = (long)NPIX * KP;
    int blocks = (int)((tot + 255) / 256);
    k_im2col<<<blocks, 256, 0, stream>>>(src, colbuf, C, KP, sStride);
    k_gemm_wmma<<<dim3(NPIX / 32, N / 64), 128, 0, stream>>>(colbuf, Wb, bias, outb, N, KP);
    k_ln<<<BB, 256, 0, stream>>>(outb, g, be, N);
  };

  const int gblocks = (int)((NG + 255) / 256);

  for (int t = 0; t < STEPS; t++) {
    for (int l = 0; l < 3; l++) {
      const float* xsrc; int xC, xs, xKP;
      if (l == 0)      { xsrc = IM + (long)t * PIX; xC = 1;  xs = TT * PIX;  xKP = 32;   }
      else if (l == 1) { xsrc = zbuf;               xC = HH; xs = PIX * HH;  xKP = 1600; }
      else             { xsrc = hbuf[1];            xC = HH; xs = PIX * HH;  xKP = 1600; }

      conv_ln(hbuf[l], HH, PIX * HH, wh_bf[l], 1600, LP(l, 5),  256, hcc, LP(l, 6),  LP(l, 7));
      conv_ln(cbuf[l], HH, PIX * HH, wc_bf[l], 1600, LP(l, 9),  192, ccc, LP(l, 10), LP(l, 11));
      conv_ln(mbuf,    HH, PIX * HH, wm_bf[l], 1600, LP(l, 13), 192, mcc, LP(l, 14), LP(l, 15));
      conv_ln(xsrc,    xC, xs,       wx_bf[l], xKP,  LP(l, 1),  448, xcc, LP(l, 2),  LP(l, 3));
      k_gates1<<<gblocks, 256, 0, stream>>>(xcc, hcc, ccc, cbuf[l], NG);

      conv_ln(cbuf[l], HH, PIX * HH, wc2m_bf[l], 1600, LP(l, 17), 256, c2m, LP(l, 18), LP(l, 19));
      k_gates2<<<gblocks, 256, 0, stream>>>(c2m, xcc, mcc, mbuf, NG);

      conv_ln(mbuf, HH, PIX * HH, wom_bf[l], 1600, LP(l, 21), 64, om, LP(l, 22), LP(l, 23));
      k_concat_bf<<<gblocks, 256, 0, stream>>>(cbuf[l], mbuf, catbuf, NG);
      k_gemm_wmma<<<dim3(NPIX / 32, 1), 128, 0, stream>>>(catbuf, wcell_bf[l], LP(l, 25),
                                                          cellb, 64, 128);
      k_gates3<<<gblocks, 256, 0, stream>>>(xcc, hcc, c2m, om, cellb, hbuf[l], NG);

      if (l == 0) {  // GHU between layer 0 and layer 1
        conv_ln(zbuf,    HH, PIX * HH, wz_bf,  1600, GP(1), 128, zcc,  GP(2), GP(3));
        conv_ln(hbuf[0], HH, PIX * HH, wxg_bf, 1600, GP(5), 128, xgcc, GP(6), GP(7));
        k_ghu<<<gblocks, 256, 0, stream>>>(zcc, xgcc, zbuf, NG);
      }
    }
    k_outconv<<<(NPIX + 255) / 256, 256, 0, stream>>>(hbuf[2], OUT_W, OUT_B, gen, t);
  }
  k_loss<<<256, 256, 0, stream>>>(gen, IM, loss);
}
